// GPT_88734024335860
// MI455X (gfx1250) — matrix operand order repaired and checked by reading, not compiled, not run on hardware
//
#include <hip/hip_runtime.h>
#include <hip/hip_bf16.h>

// ---------------------------------------------------------------------------
// Types for CDNA5 WMMA (gfx1250, wave32)
// ---------------------------------------------------------------------------
typedef __attribute__((ext_vector_type(16))) __bf16 v16bf;
typedef __attribute__((ext_vector_type(8)))  __bf16 v8bf;
typedef __attribute__((ext_vector_type(8)))  float  v8f;

__device__ __forceinline__ __bf16 f2bf(float f) {
    unsigned u = __builtin_bit_cast(unsigned, f);
    unsigned r = u + 0x7fffu + ((u >> 16) & 1u);   // round-to-nearest-even
    unsigned short h = (unsigned short)(r >> 16);
    return __builtin_bit_cast(__bf16, h);
}

__device__ __forceinline__ v8f wmma_bf16(v16bf a, v16bf b, v8f c) {
    // D = A(16x32 bf16) * B(32x16 bf16) + C(16x16 f32)
    return __builtin_amdgcn_wmma_f32_16x16x32_bf16(
        false, a, false, b, (short)0, c, false, false);
}

__device__ __forceinline__ v16bf cat8(v8bf lo, v8bf hi) {
    return __builtin_shufflevector(lo, hi, 0, 1, 2, 3, 4, 5, 6, 7,
                                           8, 9, 10, 11, 12, 13, 14, 15);
}

__device__ __forceinline__ float gelu_f(float x) {
    float x3 = x * x * x;
    return 0.5f * x * (1.0f + tanhf(0.7978845608028654f * (x + 0.044715f * x3)));
}

// ---------------------------------------------------------------------------
// Embedding: x[m,e] = wte[idx[m],e] + wpe[m % T, e]
// ---------------------------------------------------------------------------
__global__ void embed_kernel(const int* __restrict__ idx,
                             const float* __restrict__ wte,
                             const float* __restrict__ wpe,
                             float* __restrict__ x, int T, int E) {
    int m = blockIdx.x;
    int t = m % T;
    int tok = idx[m];
    const float* we = wte + (size_t)tok * E;
    const float* pe = wpe + (size_t)t * E;
    float* xr = x + (size_t)m * E;
    for (int e = threadIdx.x; e < E; e += blockDim.x) xr[e] = we[e] + pe[e];
}

// ---------------------------------------------------------------------------
// LayerNorm (fp32 in) -> bf16 out (feeds WMMA A operand)
// ---------------------------------------------------------------------------
__global__ void layernorm_kernel(const float* __restrict__ x,
                                 const float* __restrict__ w,
                                 const float* __restrict__ b,
                                 __bf16* __restrict__ out, int E) {
    int r = blockIdx.x, tid = threadIdx.x;
    const float* xr = x + (size_t)r * E;
    __shared__ float s1[256], s2[256];
    float sum = 0.f, sq = 0.f;
    for (int e = tid; e < E; e += 256) { float v = xr[e]; sum += v; sq += v * v; }
    s1[tid] = sum; s2[tid] = sq; __syncthreads();
    for (int s = 128; s > 0; s >>= 1) {
        if (tid < s) { s1[tid] += s1[tid + s]; s2[tid] += s2[tid + s]; }
        __syncthreads();
    }
    float mean = s1[0] / E;
    float var  = s2[0] / E - mean * mean;
    float rstd = rsqrtf(var + 1e-5f);
    __bf16* orow = out + (size_t)r * E;
    for (int e = tid; e < E; e += 256)
        orow[e] = f2bf((xr[e] - mean) * rstd * w[e] + b[e]);
}

// ---------------------------------------------------------------------------
// fp32 -> bf16 bulk convert (used once for wte; [V,E] is already B^T layout)
// ---------------------------------------------------------------------------
__global__ void f32_to_bf16_kernel(const float* __restrict__ in,
                                   __bf16* __restrict__ out, size_t n) {
    size_t i = (size_t)blockIdx.x * blockDim.x + threadIdx.x;
    if (i < n) out[i] = f2bf(in[i]);
}

// ---------------------------------------------------------------------------
// Weight transpose+convert: in [K,N] fp32 -> out [N,K] bf16 (K,N mult of 32).
// Makes every WMMA B-fragment a contiguous 32-byte load.
// ---------------------------------------------------------------------------
__global__ void transpose_bf16_kernel(const float* __restrict__ in,
                                      __bf16* __restrict__ out, int K, int N) {
    __shared__ float tile[32][33];
    const int kb = blockIdx.x * 32, nb = blockIdx.y * 32;
    const int tx = threadIdx.x & 31, ty = threadIdx.x >> 5;   // 32x8
#pragma unroll
    for (int i = ty; i < 32; i += 8)
        tile[i][tx] = in[(size_t)(kb + i) * N + nb + tx];
    __syncthreads();
#pragma unroll
    for (int i = ty; i < 32; i += 8)
        out[(size_t)(nb + i) * K + kb + tx] = f2bf(tile[tx][i]);
}

// ---------------------------------------------------------------------------
// WMMA GEMM:  out = epilogue(A[M,K](bf16) * Bt^T + bias),  Bt stored [N,K] bf16.
// Block = 256 threads = 8 waves; block tile 32x128; each wave: one A-frag
// shared across two 16x16 output tiles (2 WMMAs per k-step).
// Epilogue: + bias[n] -> optional GELU -> + residual[m,n] -> fp32/bf16 stores.
// ---------------------------------------------------------------------------
__global__ __launch_bounds__(256)
void gemm_wmma_kernel(const __bf16* __restrict__ A,
                      const __bf16* __restrict__ Bt,
                      const float* __restrict__ bias,
                      const float* __restrict__ res,
                      float* __restrict__ outF,
                      __bf16* __restrict__ outB,
                      int M, int N, int K, int gelu) {
    const int lane = threadIdx.x & 31;
    const int w    = threadIdx.x >> 5;
    const int m0   = blockIdx.x * 32 + (w >> 2) * 16;
    const int n0   = blockIdx.y * 128 + (w & 3) * 32;
    const int col  = lane & 15;
    const int hi   = lane >> 4;

    const int nA = n0 + col, nB = n0 + 16 + col;
    const int ncA = (nA < N) ? nA : (N - 1);
    const int ncB = (nB < N) ? nB : (N - 1);
    const int arow = m0 + col;

    const __bf16* Arow  = A  + (size_t)arow * K + hi * 16;
    const __bf16* BrowA = Bt + (size_t)ncA * K + hi * 16;
    const __bf16* BrowB = Bt + (size_t)ncB * K + hi * 16;

    v8f acc0 = {}, acc1 = {};
    for (int k = 0; k < K; k += 32) {
        // A frag: row = lane&15; K = k + hi*16 + {0..15}
        v16bf a = cat8(*(const v8bf*)(Arow + k),
                       *(const v8bf*)(Arow + k + 8));
        // B frags: col = lane&15; K = k + hi*16 + e  (contiguous 32B)
        v16bf b0 = *(const v16bf*)(BrowA + k);
        v16bf b1 = *(const v16bf*)(BrowB + k);
        acc0 = wmma_bf16(a, b0, acc0);
        acc1 = wmma_bf16(a, b1, acc1);
    }

    const float bcA = bias ? bias[ncA] : 0.f;
    const float bcB = bias ? bias[ncB] : 0.f;
    if (nA < N) {
#pragma unroll
        for (int v = 0; v < 8; ++v) {
            int r = m0 + hi * 8 + v;               // C layout: row = v + 8*hi
            float val = acc0[v] + bcA;
            if (gelu) val = gelu_f(val);
            if (res)  val += res[(size_t)r * N + nA];
            if (outF) outF[(size_t)r * N + nA] = val;
            if (outB) outB[(size_t)r * N + nA] = f2bf(val);
        }
    }
    if (nB < N) {
#pragma unroll
        for (int v = 0; v < 8; ++v) {
            int r = m0 + hi * 8 + v;
            float val = acc1[v] + bcB;
            if (gelu) val = gelu_f(val);
            if (res)  val += res[(size_t)r * N + nB];
            if (outF) outF[(size_t)r * N + nB] = val;
            if (outB) outB[(size_t)r * N + nB] = f2bf(val);
        }
    }
}

// ---------------------------------------------------------------------------
// Flash-style causal attention.  qkv: [B*T, 3E] bf16, per-head D=64.
// grid = (T/16, H, B), block = 32 (one wave per 16-row q-tile).
// S = Q K^T via WMMA; online softmax; V tile is staged memory->LDS with the
// gfx1250 async path (GLOBAL_LOAD_ASYNC_TO_LDS_B128, ASYNCcnt) so the fill
// overlaps the QK^T WMMAs and never touches VGPRs; P*V reads LDS fragments.
// ---------------------------------------------------------------------------
__global__ __launch_bounds__(32)
void attention_kernel(const __bf16* __restrict__ qkv,
                      __bf16* __restrict__ y,
                      int T, int E, int H) {
    const int qt = blockIdx.x, h = blockIdx.y, bb = blockIdx.z;
    const int lane = threadIdx.x & 31;
    const int col  = lane & 15;
    const int hi   = lane >> 4;
    const int E3   = 3 * E;

    __shared__ __bf16 P[16][32];
    __shared__ __bf16 Vt[32][64];

    // Q fragments for this q-tile (d = 0..63 -> two K-steps of 32)
    const int qrowA = qt * 16 + col;
    const __bf16* Qr = qkv + ((size_t)(bb * T + qrowA)) * E3 + h * 64 + hi * 16;
    v16bf qa0 = cat8(*(const v8bf*)(Qr),      *(const v8bf*)(Qr + 8));
    v16bf qa1 = cat8(*(const v8bf*)(Qr + 32), *(const v8bf*)(Qr + 40));

    float mrow[8], lrow[8];
#pragma unroll
    for (int v = 0; v < 8; ++v) { mrow[v] = -1e30f; lrow[v] = 0.f; }
    v8f y0 = {}, y1 = {}, y2 = {}, y3 = {};

    const int nkb = (qt * 16 + 16 + 31) >> 5;      // 32-key blocks (causal bound)
    for (int kb = 0; kb < nkb; ++kb) {
        const int key0 = kb * 32;

        // ---- async-stage V tile [32 keys x 64 d] memory -> LDS (no VGPR data)
        {
            const __bf16* vrow = qkv + (size_t)(bb * T + key0 + lane) * E3
                               + 2 * E + h * 64;
#pragma unroll
            for (int j = 0; j < 8; ++j) {
                unsigned ldsAddr = (unsigned)(size_t)&Vt[lane][j * 8];
                asm volatile("global_load_async_to_lds_b128 %0, %1, off"
                             :: "v"(ldsAddr), "v"(vrow + j * 8)
                             : "memory");
            }
        }

        // ---- S = Q K^T : b[e] = K_keyrow[kk + hi*16 + e] (contiguous)
        const __bf16* k0r = qkv + (size_t)(bb * T + key0 + col) * E3
                          + E + h * 64 + hi * 16;
        const __bf16* k1r = k0r + (size_t)16 * E3;
        v8f s0 = {}, s1 = {};
        s0 = wmma_bf16(qa0, *(const v16bf*)(k0r),      s0);
        s0 = wmma_bf16(qa1, *(const v16bf*)(k0r + 32), s0);
        s1 = wmma_bf16(qa0, *(const v16bf*)(k1r),      s1);
        s1 = wmma_bf16(qa1, *(const v16bf*)(k1r + 32), s1);

        // ---- scale, causal mask, online softmax (reductions in 16-lane halves)
        const int keyA = key0 + col, keyB = key0 + 16 + col;
#pragma unroll
        for (int v = 0; v < 8; ++v) {
            int qr = qt * 16 + 8 * hi + v;
            float a0 = s0[v] * 0.125f;             // 1/sqrt(64)
            float a1 = s1[v] * 0.125f;
            if (keyA > qr) a0 = -1e30f;
            if (keyB > qr) a1 = -1e30f;
            float mx = fmaxf(a0, a1);
            for (int off = 1; off < 16; off <<= 1)
                mx = fmaxf(mx, __shfl_xor(mx, off, 32));
            float mn = fmaxf(mrow[v], mx);
            float p0 = __expf(a0 - mn);
            float p1 = __expf(a1 - mn);
            float corr = __expf(mrow[v] - mn);
            float rs = p0 + p1;
            for (int off = 1; off < 16; off <<= 1)
                rs += __shfl_xor(rs, off, 32);
            lrow[v] = lrow[v] * corr + rs;
            mrow[v] = mn;
            y0[v] *= corr; y1[v] *= corr; y2[v] *= corr; y3[v] *= corr;
            P[8 * hi + v][col]      = f2bf(p0);
            P[8 * hi + v][16 + col] = f2bf(p1);
        }
        __syncthreads();                            // LDS write->read ordering (P)
        asm volatile("s_wait_asynccnt 0x0" ::: "memory");   // V tile landed in LDS

        // ---- P as A-fragment (row = lane&15; K = hi*16 + e)
        const __bf16* Pr = &P[col][hi * 16];
        v16bf pa;
#pragma unroll
        for (int e = 0; e < 8; ++e) { pa[e] = Pr[e]; pa[e + 8] = Pr[8 + e]; }

        // ---- y += P V ; B frag from LDS: col = d, element e -> key hi*16+e
        v16bf vb;
#pragma unroll
        for (int e = 0; e < 16; ++e) vb[e] = Vt[hi * 16 + e][col];
        y0 = wmma_bf16(pa, vb, y0);
#pragma unroll
        for (int e = 0; e < 16; ++e) vb[e] = Vt[hi * 16 + e][16 + col];
        y1 = wmma_bf16(pa, vb, y1);
#pragma unroll
        for (int e = 0; e < 16; ++e) vb[e] = Vt[hi * 16 + e][32 + col];
        y2 = wmma_bf16(pa, vb, y2);
#pragma unroll
        for (int e = 0; e < 16; ++e) vb[e] = Vt[hi * 16 + e][48 + col];
        y3 = wmma_bf16(pa, vb, y3);
        __syncthreads();                            // protect P/Vt before next iter
    }

    // ---- normalize and store y (bf16, feeds proj GEMM)
#pragma unroll
    for (int v = 0; v < 8; ++v) {
        float inv = 1.0f / lrow[v];
        int r = bb * T + qt * 16 + 8 * hi + v;
        __bf16* yo = y + (size_t)r * E + h * 64;
        yo[col]      = f2bf(y0[v] * inv);
        yo[16 + col] = f2bf(y1[v] * inv);
        yo[32 + col] = f2bf(y2[v] * inv);
        yo[48 + col] = f2bf(y3[v] * inv);
    }
}

// ---------------------------------------------------------------------------
// Per-row NLL over logits (log-softmax + gather)
// ---------------------------------------------------------------------------
__global__ void nll_kernel(const float* __restrict__ logits,
                           const int* __restrict__ tgt,
                           float* __restrict__ nll, int V) {
    int r = blockIdx.x, tid = threadIdx.x;
    const float* lr = logits + (size_t)r * V;
    __shared__ float sh[256];
    float mx = -3.4e38f;
    for (int v = tid; v < V; v += 256) mx = fmaxf(mx, lr[v]);
    sh[tid] = mx; __syncthreads();
    for (int s = 128; s > 0; s >>= 1) {
        if (tid < s) sh[tid] = fmaxf(sh[tid], sh[tid + s]);
        __syncthreads();
    }
    mx = sh[0]; __syncthreads();
    float sum = 0.f;
    for (int v = tid; v < V; v += 256) sum += __expf(lr[v] - mx);
    sh[tid] = sum; __syncthreads();
    for (int s = 128; s > 0; s >>= 1) {
        if (tid < s) sh[tid] += sh[tid + s];
        __syncthreads();
    }
    if (tid == 0) nll[r] = -(lr[tgt[r]] - mx - __logf(sh[0]));
}

__global__ void loss_kernel(const float* __restrict__ nll,
                            float* __restrict__ loss, int M) {
    __shared__ float sh[256];
    int tid = threadIdx.x;
    float s = 0.f;
    for (int i = tid; i < M; i += 256) s += nll[i];
    sh[tid] = s; __syncthreads();
    for (int st = 128; st > 0; st >>= 1) {
        if (tid < st) sh[tid] += sh[tid + st];
        __syncthreads();
    }
    if (tid == 0) *loss = sh[0] / (float)M;
}

// ---------------------------------------------------------------------------
// Host-side orchestration
// ---------------------------------------------------------------------------
extern "C" void kernel_launch(void* const* d_in, const int* in_sizes, int n_in,
                              void* d_out, int out_size, void* d_ws, size_t ws_size,
                              hipStream_t stream) {
    const int* idx      = (const int*)  d_in[0];
    const int* targets  = (const int*)  d_in[1];
    const float* wte    = (const float*)d_in[2];
    const float* wpe    = (const float*)d_in[3];
    const float* ln1_w  = (const float*)d_in[4];
    const float* ln1_b  = (const float*)d_in[5];
    const float* attn_w = (const float*)d_in[6];
    const float* attn_b = (const float*)d_in[7];
    const float* proj_w = (const float*)d_in[8];
    const float* proj_b = (const float*)d_in[9];
    const float* ln2_w  = (const float*)d_in[10];
    const float* ln2_b  = (const float*)d_in[11];
    const float* fc_w   = (const float*)d_in[12];
    const float* fc_b   = (const float*)d_in[13];
    const float* fcp_w  = (const float*)d_in[14];
    const float* fcp_b  = (const float*)d_in[15];
    const float* lnf_w  = (const float*)d_in[16];
    const float* lnf_b  = (const float*)d_in[17];

    const int E = 768, Lc = 12, Hh = 12;
    const int M = in_sizes[0];                 // B*T = 2048
    const int V = in_sizes[2] / E;             // 50257
    const int T = in_sizes[3] / E;             // 1024
    const int Bb = M / T;

    float* logits = (float*)d_out;
    float* lossp  = logits + (size_t)M * V;

    // workspace carve-out (256B aligned)
    char* p = (char*)d_ws;
    auto carve = [&](size_t bytes) {
        char* r = p; p += (bytes + 255) & ~(size_t)255; return r;
    };
    float*  x      = (float*) carve((size_t)M * E * 4);
    __bf16* xln    = (__bf16*)carve((size_t)M * E * 2);
    __bf16* qkv    = (__bf16*)carve((size_t)M * 3 * E * 2);
    __bf16* ybuf   = (__bf16*)carve((size_t)M * E * 2);
    __bf16* hbuf   = (__bf16*)carve((size_t)M * 4 * E * 2);
    __bf16* wte_bf = (__bf16*)carve((size_t)V * E * 2);
    __bf16* wt_buf = (__bf16*)carve((size_t)E * 4 * E * 2);   // reusable B^T buffer
    float*  nllbuf = (float*) carve((size_t)M * 4);
    (void)ws_size; (void)n_in; (void)out_size;

    dim3 blk(256);

    // embeddings + one-time wte bf16 conversion (wte is [V,E] == B^T layout)
    embed_kernel<<<dim3(M), blk, 0, stream>>>(idx, wte, wpe, x, T, E);
    {
        size_t n = (size_t)V * E;
        f32_to_bf16_kernel<<<dim3((unsigned)((n + 255) / 256)), blk, 0, stream>>>(wte, wte_bf, n);
    }

    auto xpose = [&](const float* W, int Kk, int Nn) {   // [K,N] f32 -> [N,K] bf16
        transpose_bf16_kernel<<<dim3(Kk / 32, Nn / 32), blk, 0, stream>>>(W, wt_buf, Kk, Nn);
    };
    auto gemm = [&](const __bf16* A, const __bf16* Bt, const float* bias,
                    const float* res, float* oF, __bf16* oB,
                    int Mm, int Nn, int Kk, int gelu) {
        dim3 g(Mm / 32, (Nn + 127) / 128);
        gemm_wmma_kernel<<<g, blk, 0, stream>>>(A, Bt, bias, res, oF, oB, Mm, Nn, Kk, gelu);
    };

    for (int l = 0; l < Lc; ++l) {
        const float* aw  = attn_w + (size_t)l * E * 3 * E;
        const float* ab  = attn_b + (size_t)l * 3 * E;
        const float* pw  = proj_w + (size_t)l * E * E;
        const float* pb  = proj_b + (size_t)l * E;
        const float* fw  = fc_w   + (size_t)l * E * 4 * E;
        const float* fb  = fc_b   + (size_t)l * 4 * E;
        const float* fpw = fcp_w  + (size_t)l * 4 * E * E;
        const float* fpb = fcp_b  + (size_t)l * E;

        layernorm_kernel<<<dim3(M), blk, 0, stream>>>(x, ln1_w + l * E, ln1_b + l * E, xln, E);
        xpose(aw, E, 3 * E);
        gemm(xln, wt_buf, ab, nullptr, nullptr, qkv, M, 3 * E, E, 0);      // QKV
        attention_kernel<<<dim3(T / 16, Hh, Bb), dim3(32), 0, stream>>>(qkv, ybuf, T, E, Hh);
        xpose(pw, E, E);
        gemm(ybuf, wt_buf, pb, x, x, nullptr, M, E, E, 0);                 // proj + residual
        layernorm_kernel<<<dim3(M), blk, 0, stream>>>(x, ln2_w + l * E, ln2_b + l * E, xln, E);
        xpose(fw, E, 4 * E);
        gemm(xln, wt_buf, fb, nullptr, nullptr, hbuf, M, 4 * E, E, 1);     // fc + GELU
        xpose(fpw, 4 * E, E);
        gemm(hbuf, wt_buf, fpb, x, x, nullptr, M, E, 4 * E, 0);            // fcproj + residual
    }

    // final LN + tied lm_head (wte_bf is already [N,K]) + loss
    layernorm_kernel<<<dim3(M), blk, 0, stream>>>(x, lnf_w, lnf_b, xln, E);
    gemm(xln, wte_bf, nullptr, nullptr, logits, nullptr, M, V, E, 0);
    nll_kernel<<<dim3(M), blk, 0, stream>>>(logits, targets, nllbuf, V);
    loss_kernel<<<dim3(1), blk, 0, stream>>>(nllbuf, lossp, M);
}
